// Antenna_message_generate_36867999269161
// MI455X (gfx1250) — matrix operand order repaired and checked by reading, not compiled, not run on hardware
//
#include <hip/hip_runtime.h>
#include <hip/hip_bf16.h>
#include <math.h>

typedef __attribute__((ext_vector_type(16))) _Float16 v16h;
typedef __attribute__((ext_vector_type(8)))  _Float16 v8h;
typedef __attribute__((ext_vector_type(8)))  float    v8f;

// Problem sizes (from reference)
constexpr int BATCH = 2048;
constexpr int H1_   = 512;
constexpr int H2_   = 512;
constexpr int M_    = 16;
constexpr int KIN   = 64;   // 48 real input features zero-padded to 64 (K multiple of 32)
constexpr int ROWS  = 64;   // rows per block = R*U (one batch element)
constexpr int XS_STRIDE = 72;   // padded LDS row stride (f16 units) for x
constexpr int HS_STRIDE = 520;  // padded LDS row stride (f16 units) for h1/h2

// ---------------- weight conversion: f32 -> f16 into workspace ----------------
__global__ __launch_bounds__(256) void prep_weights(
    const float* __restrict__ W1, const float* __restrict__ W2, const float* __restrict__ W3,
    _Float16* __restrict__ W1h, _Float16* __restrict__ W2h, _Float16* __restrict__ W3h)
{
    int i = blockIdx.x * 256 + threadIdx.x;
    if (i < KIN * H1_) {
        int k = i >> 9, n = i & 511;
        W1h[i] = (_Float16)(k < 48 ? W1[k * H1_ + n] : 0.0f);
    } else if (i < KIN * H1_ + H1_ * H2_) {
        int j = i - KIN * H1_;
        W2h[j] = (_Float16)W2[j];
    } else if (i < KIN * H1_ + H1_ * H2_ + H2_ * M_) {
        int j = i - (KIN * H1_ + H1_ * H2_);
        W3h[j] = (_Float16)W3[j];
    }
}

// ---------------- fragment loaders ----------------
// A layout (16-bit A 16x32, ISA 7.12.2): lane L<16 -> M=L, K = [k0+0..7] and [k0+16..23];
// lane L>=16 -> M=L-16, K = [k0+8..15] and [k0+24..31]. Two 16B chunks per lane.
__device__ __forceinline__ v16h load_a_frag(const _Float16* base, int stride, int lane, int k0)
{
    int m  = lane & 15;
    int hl = lane >> 4;
    const _Float16* p = base + m * stride + k0 + 8 * hl;
    v8h lo = *(const v8h*)p;
    v8h hi = *(const v8h*)(p + 16);
    v16h a;
#pragma unroll
    for (int i = 0; i < 8; ++i) { a[i] = lo[i]; a[8 + i] = hi[i]; }
    return a;
}

// B layout (K x 16): lane L holds column N=n0+L%16, K = [k0+8h..+7] and [k0+16+8h..+7], h=L/16.
__device__ __forceinline__ v16h load_b_frag(const _Float16* W, int ncols, int lane, int k0, int n0)
{
    const _Float16* p = W + (size_t)(k0 + 8 * (lane >> 4)) * ncols + n0 + (lane & 15);
    v16h b;
#pragma unroll
    for (int i = 0; i < 8; ++i) { b[i] = p[(size_t)i * ncols]; b[8 + i] = p[(size_t)(16 + i) * ncols]; }
    return b;
}

// ---------------- fused gather + 3-layer MLP ----------------
// Block: 256 threads = 8 waves, one batch element b (64 rows).
// Waves: wave w owns a unique 64-column N-slice (n0 = 64*w) and ALL 4 M-tiles
// (4x4 tile grid per wave) -> every W-fragment is read exactly once per block.
__global__ __launch_bounds__(256) void mlp_fused(
    const float* __restrict__ Cin, const float* __restrict__ Fin, const float* __restrict__ Hin,
    const _Float16* __restrict__ W1h, const float* __restrict__ b1,
    const _Float16* __restrict__ W2h, const float* __restrict__ b2,
    const _Float16* __restrict__ W3h, const float* __restrict__ b3,
    float* __restrict__ out)
{
    __shared__ __align__(16) _Float16 xs [ROWS * XS_STRIDE];
    __shared__ __align__(16) _Float16 h1s[ROWS * HS_STRIDE];
    __shared__ __align__(16) _Float16 h2s[ROWS * HS_STRIDE];

    const int b    = blockIdx.x;
    const int t    = threadIdx.x;
    const int lane = t & 31;
    const int wave = t >> 5;     // 0..7 : N-slice of 64 columns
    const int col  = lane & 15;
    const int hl   = lane >> 4;
    const int n0w  = wave * 64;

    // ---- gather x = [F(16) | sum_u C(16) | h_kr(8) | h_ki(8) | pad(16 zeros)] ----
    {
        int row = t >> 2;          // 0..63, row = r*8 + u
        int q   = t & 3;           // 16-feature chunk
        int r   = row >> 3;
        int u   = row & 7;
        float vals[16];
        if (q == 0) {
#pragma unroll
            for (int j = 0; j < 16; ++j) vals[j] = Fin[(b * 16 + j) * 8 + r];
        } else if (q == 1) {
#pragma unroll
            for (int j = 0; j < 16; ++j) {
                float s = 0.f;
#pragma unroll
                for (int uu = 0; uu < 8; ++uu)
                    s += Cin[(((b * 8 + uu) * 8 + r) << 4) + j];
                vals[j] = s;
            }
        } else if (q == 2) {
            const float* Hp = Hin + (b * 8 + u) * 128 + r * 8;
#pragma unroll
            for (int j = 0; j < 8; ++j) { vals[j] = Hp[j]; vals[8 + j] = Hp[64 + j]; }
        } else {
#pragma unroll
            for (int j = 0; j < 16; ++j) vals[j] = 0.f;
        }
        v8h lo, hi;
#pragma unroll
        for (int j = 0; j < 8; ++j) { lo[j] = (_Float16)vals[j]; hi[j] = (_Float16)vals[8 + j]; }
        *(v8h*)&xs[row * XS_STRIDE + q * 16]     = lo;
        *(v8h*)&xs[row * XS_STRIDE + q * 16 + 8] = hi;
    }
    __syncthreads();

    // ---- layer 1: [64x64] @ [64x512] -> relu -> h1s (f16) ----
    {
        v8f acc[4][4];
#pragma unroll
        for (int nt = 0; nt < 4; ++nt) {
            float bv = b1[n0w + nt * 16 + col];
#pragma unroll
            for (int mt = 0; mt < 4; ++mt)
#pragma unroll
                for (int i = 0; i < 8; ++i) acc[mt][nt][i] = bv;
        }
#pragma unroll
        for (int k0 = 0; k0 < KIN; k0 += 32) {
            v16h a[4];
#pragma unroll
            for (int mt = 0; mt < 4; ++mt)
                a[mt] = load_a_frag(&xs[16 * mt * XS_STRIDE], XS_STRIDE, lane, k0);
#pragma unroll
            for (int nt = 0; nt < 4; ++nt) {
                v16h bf = load_b_frag(W1h, H1_, lane, k0, n0w + nt * 16);
#pragma unroll
                for (int mt = 0; mt < 4; ++mt)
                    acc[mt][nt] = __builtin_amdgcn_wmma_f32_16x16x32_f16(
                        false, a[mt], false, bf, (short)0, acc[mt][nt], false, false);
            }
        }
#pragma unroll
        for (int mt = 0; mt < 4; ++mt)
#pragma unroll
            for (int nt = 0; nt < 4; ++nt)
#pragma unroll
                for (int rr = 0; rr < 8; ++rr) {
                    float v = fmaxf(acc[mt][nt][rr], 0.f);
                    h1s[(16 * mt + rr + 8 * hl) * HS_STRIDE + n0w + nt * 16 + col] = (_Float16)v;
                }
    }
    __syncthreads();

    // ---- layer 2: [64x512] @ [512x512] -> relu -> h2s (f16) ----
    {
        v8f acc[4][4];
#pragma unroll
        for (int nt = 0; nt < 4; ++nt) {
            float bv = b2[n0w + nt * 16 + col];
#pragma unroll
            for (int mt = 0; mt < 4; ++mt)
#pragma unroll
                for (int i = 0; i < 8; ++i) acc[mt][nt][i] = bv;
        }
        for (int k0 = 0; k0 < H1_; k0 += 32) {
            v16h a[4];
#pragma unroll
            for (int mt = 0; mt < 4; ++mt)
                a[mt] = load_a_frag(&h1s[16 * mt * HS_STRIDE], HS_STRIDE, lane, k0);
            if (k0 + 32 < H1_)  // prefetch next W2 k-slice (-> global_prefetch_b8)
                __builtin_prefetch(W2h + (size_t)(k0 + 32 + lane) * H2_ + n0w, 0, 0);
#pragma unroll
            for (int nt = 0; nt < 4; ++nt) {
                v16h bf = load_b_frag(W2h, H2_, lane, k0, n0w + nt * 16);
#pragma unroll
                for (int mt = 0; mt < 4; ++mt)
                    acc[mt][nt] = __builtin_amdgcn_wmma_f32_16x16x32_f16(
                        false, a[mt], false, bf, (short)0, acc[mt][nt], false, false);
            }
        }
#pragma unroll
        for (int mt = 0; mt < 4; ++mt)
#pragma unroll
            for (int nt = 0; nt < 4; ++nt)
#pragma unroll
                for (int rr = 0; rr < 8; ++rr) {
                    float v = fmaxf(acc[mt][nt][rr], 0.f);
                    h2s[(16 * mt + rr + 8 * hl) * HS_STRIDE + n0w + nt * 16 + col] = (_Float16)v;
                }
    }
    __syncthreads();

    // ---- layer 3: [64x512] @ [512x16] -> tanh -> out (waves 0..3, one M-tile each) ----
    if (wave < 4) {
        const int mt = wave;
        v8f acc;
        {
            float bv = b3[col];
#pragma unroll
            for (int i = 0; i < 8; ++i) acc[i] = bv;
        }
        for (int k0 = 0; k0 < H2_; k0 += 32) {
            v16h a  = load_a_frag(&h2s[16 * mt * HS_STRIDE], HS_STRIDE, lane, k0);
            v16h bf = load_b_frag(W3h, M_, lane, k0, 0);
            acc = __builtin_amdgcn_wmma_f32_16x16x32_f16(
                false, a, false, bf, (short)0, acc, false, false);
        }
        float* op = out + (size_t)b * (ROWS * M_);
#pragma unroll
        for (int rr = 0; rr < 8; ++rr) {
            int mrow = 16 * mt + rr + 8 * hl;
            op[mrow * M_ + col] = tanhf(acc[rr]);
        }
    }
}

// ---------------- launch ----------------
extern "C" void kernel_launch(void* const* d_in, const int* in_sizes, int n_in,
                              void* d_out, int out_size, void* d_ws, size_t ws_size,
                              hipStream_t stream)
{
    const float* C  = (const float*)d_in[0];
    const float* F  = (const float*)d_in[1];
    const float* H  = (const float*)d_in[2];
    const float* W1 = (const float*)d_in[3];
    const float* b1 = (const float*)d_in[4];
    const float* W2 = (const float*)d_in[5];
    const float* b2 = (const float*)d_in[6];
    const float* W3 = (const float*)d_in[7];
    const float* b3 = (const float*)d_in[8];

    _Float16* W1h = (_Float16*)d_ws;                 // [64][512]
    _Float16* W2h = W1h + KIN * H1_;                 // [512][512]
    _Float16* W3h = W2h + H1_ * H2_;                 // [512][16]

    int total = KIN * H1_ + H1_ * H2_ + H2_ * M_;
    prep_weights<<<(total + 255) / 256, 256, 0, stream>>>(W1, W2, W3, W1h, W2h, W3h);

    mlp_fused<<<BATCH, 256, 0, stream>>>(C, F, H, W1h, b1, W2h, b2, W3h, b3, (float*)d_out);
}
